// ProjectedAttentionLayer_58755152610005
// MI455X (gfx1250) — compile-verified, matrix operand order checked by reading, not run on hardware
//
#include <hip/hip_runtime.h>

typedef __attribute__((ext_vector_type(16))) __bf16 v16bf;
typedef __attribute__((ext_vector_type(4)))  __bf16 v4bf;
typedef __attribute__((ext_vector_type(8)))  float  v8f;
typedef __attribute__((__vector_size__(16))) int    v4i;

// ---------------------------------------------------------------------------
// CDNA5 async global->LDS copy (GLOBAL_LOAD_ASYNC_TO_LDS_B128, ASYNCcnt),
// guarded so absence of the builtin falls back to a plain 16B copy.
// Probe-derived signature: param0 = AS1 v4i* (global src), param1 = AS3 v4i*.
// ---------------------------------------------------------------------------
#if defined(__AMDGCN__) && __has_builtin(__builtin_amdgcn_global_load_async_to_lds_b128)
#define HAVE_ASYNC_LDS 1
#else
#define HAVE_ASYNC_LDS 0
#endif

__device__ __forceinline__ void copy16B_g2l(const __bf16* __restrict__ gsrc,
                                            __bf16* __restrict__ ldst)
{
#if HAVE_ASYNC_LDS
    __builtin_amdgcn_global_load_async_to_lds_b128(
        (__attribute__((address_space(1))) v4i*)gsrc,
        (__attribute__((address_space(3))) v4i*)ldst,
        /*offset=*/0, /*cpol=*/0);
#else
    *(uint4*)ldst = *(const uint4*)gsrc;
#endif
}

__device__ __forceinline__ void async_wait0()
{
#if HAVE_ASYNC_LDS
#if __has_builtin(__builtin_amdgcn_s_wait_asynccnt)
    __builtin_amdgcn_s_wait_asynccnt(0);
#else
    asm volatile("s_wait_asynccnt 0x0" ::: "memory");
#endif
#endif
}

// ---------------------------------------------------------------------------
// Problem constants
// ---------------------------------------------------------------------------
constexpr int BATCH = 4;
constexpr int NQn   = 2048;
constexpr int NKn   = 2048;
constexpr int Dn    = 1024;

// GEMM tiling
constexpr int BM = 128, BN = 128, BK = 32;
constexpr int LDSK = 40;   // padded K-stride in bf16 units: 80B row pitch,
                           // 16B aligned for b128, conflict-free (gcd(20,64)=4)

// ---------------------------------------------------------------------------
// Tiled transpose + convert-to-bf16:  out[c][r] = (bf16) in[r][c]
// ---------------------------------------------------------------------------
template <typename InT>
__global__ __launch_bounds__(256) void transpose_to_bf16(
    const InT* __restrict__ in, __bf16* __restrict__ out,
    int rows, int cols, long long sIn, long long sOut)
{
    __shared__ float tile[32][33];
    const InT* ip = in  + (size_t)blockIdx.z * sIn;
    __bf16*    op = out + (size_t)blockIdx.z * sOut;
    const int c0 = blockIdx.x * 32;
    const int r0 = blockIdx.y * 32;
    const int x = threadIdx.x, y = threadIdx.y;
    #pragma unroll
    for (int j = 0; j < 32; j += 8)
        tile[y + j][x] = (float)ip[(long long)(r0 + y + j) * cols + (c0 + x)];
    __syncthreads();
    #pragma unroll
    for (int j = 0; j < 32; j += 8)
        op[(long long)(c0 + y + j) * rows + (r0 + x)] = (__bf16)tile[x][y + j];
}

// ---------------------------------------------------------------------------
// WMMA GEMM:  C[m][n] = scale * sum_k A[m][k] * Bt[n][k]  (+ bias[n])
//   A  : (batch) M x K row-major (float or bf16), row stride lda
//   Bt : (batch) N x K row-major bf16 (i.e. B^T), row stride ldb
//   C  : (batch) M x N, row stride ldc, bf16 or f32
// grid = (N/BN, M/BM, batch), block = 256 (8 waves of 32)
// Double-buffered LDS; bf16 tiles staged via async global->LDS when available.
// ---------------------------------------------------------------------------
template <typename AT, bool OUT_BF16, bool HAS_BIAS>
__global__ __launch_bounds__(256) void gemm_wmma_bf16(
    const AT* __restrict__ A, long long strideA,
    const __bf16* __restrict__ Bt, long long strideB,
    void* __restrict__ Cv, long long strideC,
    const float* __restrict__ bias,
    int K, int lda, int ldb, int ldc, float scale)
{
    __shared__ __bf16 sA[2][BM * LDSK];
    __shared__ __bf16 sB[2][BN * LDSK];

    const int tid   = threadIdx.x;
    const int rowM0 = blockIdx.y * BM;
    const int colN0 = blockIdx.x * BN;
    const AT*     Ab = A  + (size_t)blockIdx.z * strideA;
    const __bf16* Bb = Bt + (size_t)blockIdx.z * strideB;

    const int lane  = tid & 31;
    const int wave  = tid >> 5;
    const int wm    = wave & 3;   // 4 wave rows  -> 32 M each
    const int wn    = wave >> 2;  // 2 wave cols  -> 64 N each
    const int lrow  = lane & 15;
    const int khalf = lane >> 4;

    const int srow  = tid >> 1;   // staging row 0..127
    const int shalf = tid & 1;    // staging 16-element K half

    v8f acc[2][4];
    {
        v8f z = {0.f, 0.f, 0.f, 0.f, 0.f, 0.f, 0.f, 0.f};
        #pragma unroll
        for (int i = 0; i < 2; i++)
            #pragma unroll
            for (int j = 0; j < 4; j++) acc[i][j] = z;
    }

    // ---- staging helpers ----
    auto stageA = [&](int buf, int k0) {
        if constexpr (sizeof(AT) == 4) {
            // f32 -> bf16 convert path (synchronous)
            const float4* src = (const float4*)(Ab + (long long)(rowM0 + srow) * lda + k0 + shalf * 16);
            v4bf* dst = (v4bf*)(sA[buf] + srow * LDSK + shalf * 16);
            #pragma unroll
            for (int j = 0; j < 4; j++) {
                float4 f = src[j];
                v4bf o;
                o[0] = (__bf16)f.x; o[1] = (__bf16)f.y;
                o[2] = (__bf16)f.z; o[3] = (__bf16)f.w;
                dst[j] = o;
            }
        } else {
            const __bf16* src = Ab + (long long)(rowM0 + srow) * lda + k0 + shalf * 16;
            __bf16* dst = sA[buf] + srow * LDSK + shalf * 16;
            copy16B_g2l(src,     dst);
            copy16B_g2l(src + 8, dst + 8);
        }
    };
    auto stageB = [&](int buf, int k0) {
        const __bf16* src = Bb + (long long)(colN0 + srow) * ldb + k0 + shalf * 16;
        __bf16* dst = sB[buf] + srow * LDSK + shalf * 16;
        copy16B_g2l(src,     dst);
        copy16B_g2l(src + 8, dst + 8);
    };

    // ---- prologue: fill buffer 0 ----
    stageA(0, 0);
    stageB(0, 0);
    async_wait0();
    __syncthreads();

    int cur = 0;
    for (int k0 = 0; k0 < K; k0 += BK) {
        // kick off next tile into the other buffer while computing this one
        if (k0 + BK < K) {
            stageA(cur ^ 1, k0 + BK);
            stageB(cur ^ 1, k0 + BK);
#if !HAVE_ASYNC_LDS
            if (k0 + 2 * BK < K) {
                __builtin_prefetch(Ab + (long long)(rowM0 + srow) * lda + k0 + 2 * BK, 0, 2);
                __builtin_prefetch(Bb + (long long)(colN0 + srow) * ldb + k0 + 2 * BK, 0, 2);
            }
#endif
        }

        // ---- fragments per the 16-bit 16x32 A/B VGPR layout ----
        const __bf16* a_base = sA[cur];
        const __bf16* b_base = sB[cur];
        v16bf afrag[2], bfrag[4];
        #pragma unroll
        for (int mt = 0; mt < 2; mt++) {
            const uint4* p = (const uint4*)(a_base + (wm * 32 + mt * 16 + lrow) * LDSK);
            union { v16bf v; uint4 q[2]; } u;
            u.q[0] = p[khalf];
            u.q[1] = p[2 + khalf];
            afrag[mt] = u.v;
        }
        #pragma unroll
        for (int nt = 0; nt < 4; nt++) {
            const uint4* p = (const uint4*)(b_base + (wn * 64 + nt * 16 + lrow) * LDSK);
            union { v16bf v; uint4 q[2]; } u;
            u.q[0] = p[khalf];
            u.q[1] = p[2 + khalf];
            bfrag[nt] = u.v;
        }

        // ---- 8 WMMAs: 32x64 per wave per K-step ----
        #pragma unroll
        for (int mt = 0; mt < 2; mt++)
            #pragma unroll
            for (int nt = 0; nt < 4; nt++)
                acc[mt][nt] = __builtin_amdgcn_wmma_f32_16x16x32_bf16(
                    false, afrag[mt], false, bfrag[nt],
                    (short)0, acc[mt][nt], false, false);

        async_wait0();     // my writes to the next buffer have landed in LDS
        __syncthreads();   // everyone done reading cur / writing next
        cur ^= 1;
    }

    // ---- epilogue: C layout = VGPR i -> M = i + 8*khalf, N = lrow ----
    const long long cbase = (long long)blockIdx.z * strideC;
    #pragma unroll
    for (int mt = 0; mt < 2; mt++) {
        #pragma unroll
        for (int nt = 0; nt < 4; nt++) {
            const int n = colN0 + wn * 64 + nt * 16 + lrow;
            const float badd = HAS_BIAS ? bias[n] : 0.0f;
            #pragma unroll
            for (int i = 0; i < 8; i++) {
                const int m = rowM0 + wm * 32 + mt * 16 + i + 8 * khalf;
                const float val = acc[mt][nt][i] * scale + badd;
                if constexpr (OUT_BF16)
                    ((__bf16*)Cv)[cbase + (long long)m * ldc + n] = (__bf16)val;
                else
                    ((float*)Cv)[cbase + (long long)m * ldc + n] = val;
            }
        }
    }
}

// ---------------------------------------------------------------------------
// In-place row softmax on bf16 scores. One block per row; f32 math.
// ---------------------------------------------------------------------------
__global__ __launch_bounds__(256) void softmax_rows_bf16(__bf16* __restrict__ S, int n)
{
    __shared__ float red[256];
    const int t = threadIdx.x;
    __bf16* p = S + (long long)blockIdx.x * n;
    const int PER = n / 256;   // 8 for n = 2048

    float v[8];
    float mx = -3.0e38f;
    #pragma unroll
    for (int j = 0; j < 8; j++) {
        v[j] = (j < PER) ? (float)p[t * PER + j] : -3.0e38f;
        mx = fmaxf(mx, v[j]);
    }
    red[t] = mx;
    __syncthreads();
    for (int s = 128; s > 0; s >>= 1) {
        if (t < s) red[t] = fmaxf(red[t], red[t + s]);
        __syncthreads();
    }
    mx = red[0];
    __syncthreads();

    float sum = 0.0f;
    #pragma unroll
    for (int j = 0; j < 8; j++) {
        if (j < PER) {
            v[j] = __expf(v[j] - mx);
            sum += v[j];
        }
    }
    red[t] = sum;
    __syncthreads();
    for (int s = 128; s > 0; s >>= 1) {
        if (t < s) red[t] += red[t + s];
        __syncthreads();
    }
    const float inv = 1.0f / red[0];
    #pragma unroll
    for (int j = 0; j < 8; j++)
        if (j < PER) p[t * PER + j] = (__bf16)(v[j] * inv);
}

// ---------------------------------------------------------------------------
// Host orchestration
// ---------------------------------------------------------------------------
extern "C" void kernel_launch(void* const* d_in, const int* in_sizes, int n_in,
                              void* d_out, int out_size, void* d_ws, size_t ws_size,
                              hipStream_t stream)
{
    (void)in_sizes; (void)n_in; (void)out_size; (void)ws_size;

    const float* queries = (const float*)d_in[0];
    const float* keys    = (const float*)d_in[1];
    const float* values  = (const float*)d_in[2];
    const float* Wq      = (const float*)d_in[3];
    const float* bq      = (const float*)d_in[4];
    const float* Wk      = (const float*)d_in[5];
    const float* bk      = (const float*)d_in[6];
    const float* Wv      = (const float*)d_in[7];
    const float* bv      = (const float*)d_in[8];
    float* out = (float*)d_out;

    const long long MPROJ = (long long)BATCH * NQn;   // 8192

    // workspace carve-up (~124 MB total)
    char* wsp = (char*)d_ws;
    size_t off = 0;
    auto take = [&](size_t bytes) -> char* {
        char* p = wsp + off;
        off += (bytes + 255) & ~(size_t)255;
        return p;
    };
    __bf16* qb   = (__bf16*)take((size_t)MPROJ * Dn * 2);
    __bf16* kb   = (__bf16*)take((size_t)MPROJ * Dn * 2);
    __bf16* v1b  = (__bf16*)take((size_t)MPROJ * Dn * 2);
    __bf16* v2b  = (__bf16*)take((size_t)MPROJ * Dn * 2);
    __bf16* v2tb = (__bf16*)take((size_t)BATCH * Dn * NKn * 2);
    __bf16* Wqt  = (__bf16*)take((size_t)Dn * Dn * 2);
    __bf16* Wkt  = (__bf16*)take((size_t)Dn * Dn * 2);
    __bf16* Wvt  = (__bf16*)take((size_t)Dn * Dn * 2);
    __bf16* Sb   = (__bf16*)take((size_t)BATCH * NQn * NKn * 2);

    const dim3 tb(32, 8);
    const dim3 gb(256);
    const float inv_sqrt_d = 0.03125f;  // 1/sqrt(1024)

    // 1) weights -> bf16, transposed (N-major x K-contiguous B operands)
    transpose_to_bf16<float><<<dim3(Dn / 32, Dn / 32, 1), tb, 0, stream>>>(Wq, Wqt, Dn, Dn, 0, 0);
    transpose_to_bf16<float><<<dim3(Dn / 32, Dn / 32, 1), tb, 0, stream>>>(Wk, Wkt, Dn, Dn, 0, 0);
    transpose_to_bf16<float><<<dim3(Dn / 32, Dn / 32, 1), tb, 0, stream>>>(Wv, Wvt, Dn, Dn, 0, 0);

    // 2) projections: q, k, v1, v2 (value projection applied twice, per reference)
    gemm_wmma_bf16<float, true, true><<<dim3(Dn / BN, MPROJ / BM, 1), gb, 0, stream>>>(
        queries, 0, Wqt, 0, qb, 0, bq, Dn, Dn, Dn, Dn, 1.0f);
    gemm_wmma_bf16<float, true, true><<<dim3(Dn / BN, MPROJ / BM, 1), gb, 0, stream>>>(
        keys, 0, Wkt, 0, kb, 0, bk, Dn, Dn, Dn, Dn, 1.0f);
    gemm_wmma_bf16<float, true, true><<<dim3(Dn / BN, MPROJ / BM, 1), gb, 0, stream>>>(
        values, 0, Wvt, 0, v1b, 0, bv, Dn, Dn, Dn, Dn, 1.0f);
    gemm_wmma_bf16<__bf16, true, true><<<dim3(Dn / BN, MPROJ / BM, 1), gb, 0, stream>>>(
        v1b, 0, Wvt, 0, v2b, 0, bv, Dn, Dn, Dn, Dn, 1.0f);

    // 3) v2 -> v2^T per batch (B operand for O = P * v2)
    transpose_to_bf16<__bf16><<<dim3(Dn / 32, NKn / 32, BATCH), tb, 0, stream>>>(
        v2b, v2tb, NKn, Dn, (long long)NKn * Dn, (long long)Dn * NKn);

    // 4) scores = q * k^T / sqrt(D)   (k already N-major x K-contiguous)
    gemm_wmma_bf16<__bf16, true, false><<<dim3(NKn / BN, NQn / BM, BATCH), gb, 0, stream>>>(
        qb, (long long)NQn * Dn, kb, (long long)NKn * Dn, Sb, (long long)NQn * NKn,
        nullptr, Dn, Dn, Dn, NKn, inv_sqrt_d);

    // 5) softmax over NK, in place
    softmax_rows_bf16<<<dim3(BATCH * NQn), gb, 0, stream>>>(Sb, NKn);

    // 6) out = P * v2   (f32 straight to d_out)
    gemm_wmma_bf16<__bf16, false, false><<<dim3(Dn / BN, NQn / BM, BATCH), gb, 0, stream>>>(
        Sb, (long long)NQn * NKn, v2tb, (long long)Dn * NKn, out, (long long)NQn * Dn,
        nullptr, NKn, NKn, NKn, Dn, 1.0f);
}
